// CA_66486093742236
// MI455X (gfx1250) — matrix operand order repaired and checked by reading, not compiled, not run on hardware
//
#include <hip/hip_runtime.h>

#define CC   64
#define CIN_ 256
#define NN   4096
#define BB   8

typedef __attribute__((ext_vector_type(16))) _Float16 v16h;
typedef __attribute__((ext_vector_type(8)))  _Float16 v8h;
typedef __attribute__((ext_vector_type(8)))  float    v8f;
typedef __attribute__((ext_vector_type(4)))  unsigned int v4u;
typedef __attribute__((ext_vector_type(8)))  int      v8i;
typedef __attribute__((ext_vector_type(4)))  int      v4i;

// ---- WMMA fragment loaders (wave32; contiguous k order, same for A and B) ----
// A (16x32 f16): lane holds row M=lane%16; elems 0..15 -> K = base..base+15,
// where base = kstep*32 + 16*(lane/16).
__device__ __forceinline__ v16h ld_frag_a(const _Float16* __restrict__ row, int base) {
  v8h lo = *(const v8h*)(row + base);
  v8h hi = *(const v8h*)(row + base + 8);
  return __builtin_shufflevector(lo, hi, 0,1,2,3,4,5,6,7,8,9,10,11,12,13,14,15);
}
// B (32x16 f16): lane holds col N=lane%16; elems 0..15 -> K = base..base+15,
// base = kstep*32 + 16*(lane/16). Needs [N][K] (transposed) storage.
__device__ __forceinline__ v16h ld_frag_b(const _Float16* __restrict__ row, int base) {
  v8h lo = *(const v8h*)(row + base);
  v8h hi = *(const v8h*)(row + base + 8);
  return __builtin_shufflevector(lo, hi, 0,1,2,3,4,5,6,7,8,9,10,11,12,13,14,15);
}

__device__ __forceinline__ v8f wmma16(v16h a, v16h b, v8f c) {
  return __builtin_amdgcn_wmma_f32_16x16x32_f16(false, a, false, b, (short)0, c, false, false);
}

// ---- Tensor Data Mover: 2-D tile (f16) global -> LDS, with row padding ----
// data_size=2B; pad_enable: after every 32 DWORDs (one 64-half row = 128B)
// skip 4 DWORDs (8 halves) -> matches the [rows][72] padded LDS layout.
// clang-23 toolchain: 6-arg builtin (g0 v4u, g1 v8i, g2 v4i, g3 v4i, v8i, cpol).
__device__ __forceinline__ void tdm_load_2d_f16(const void* gptr, unsigned lds_addr,
                                                unsigned d0, unsigned d1,
                                                unsigned stride0,
                                                unsigned tile_d0, unsigned tile_d1)
{
  unsigned long long ga = (unsigned long long)(uintptr_t)gptr;
  v4u g0;
  g0.x = 1u;                                         // count=1 (valid), user mode
  g0.y = lds_addr;                                   // LDS byte address
  g0.z = (unsigned)ga;                               // global addr [31:0]
  g0.w = (unsigned)((ga >> 32) & 0x01FFFFFFu) | (2u << 30);  // addr[56:32] | type=2
  v8i g1;
  g1[0] = (int)((1u << 16)      // data_size = 2 bytes
              | (1u << 20)      // pad_enable
              | (4u << 22)      // pad_interval: 32 DWORDs
              | (3u << 25));    // pad_amount:   4 DWORDs
  g1[1] = (int)((d0 & 0xFFFFu) << 16);                       // tensor_dim0 lo
  g1[2] = (int)(((d0 >> 16) & 0xFFFFu) | ((d1 & 0xFFFFu) << 16));
  g1[3] = (int)(((d1 >> 16) & 0xFFFFu) | ((tile_d0 & 0xFFFFu) << 16));
  g1[4] = (int)(tile_d1 & 0xFFFFu);                          // tile_dim1, tile_dim2=0
  g1[5] = (int)stride0;                                      // tensor_dim0_stride lo
  g1[6] = 0;
  g1[7] = 0;
  v4i z4 = {0, 0, 0, 0};                                     // 2-D: groups 2/3 unused
  v8i z8 = {0, 0, 0, 0, 0, 0, 0, 0};
  __builtin_amdgcn_tensor_load_to_lds(g0, g1, z4, z4, z8, 0);
}

// =====================================================================
// Kernel 1: xproj = convert_w @ x (+b), then q,k,v projections in f16.
//  Layouts produced:  xproj fp32 [B][C][N] ; qT,kT f16 [B][N][C] ; vC f16 [B][C][N]
// =====================================================================
__global__ __launch_bounds__(128)
void ca_proj_kernel(const float* __restrict__ x,  const float* __restrict__ cw,
                    const float* __restrict__ cb,
                    const float* __restrict__ qw, const float* __restrict__ qb,
                    const float* __restrict__ kw, const float* __restrict__ kb,
                    const float* __restrict__ vw, const float* __restrict__ vb,
                    float* __restrict__ xproj, _Float16* __restrict__ qT,
                    _Float16* __restrict__ kT, _Float16* __restrict__ vC)
{
  __shared__ __align__(16) _Float16 cw_s[64][72];   // [c][ci_local]
  __shared__ __align__(16) _Float16 xT_s[64][72];   // [n_local][ci_local]
  __shared__ __align__(16) _Float16 wq_s[64][72];   // [o][c]
  __shared__ __align__(16) _Float16 wk_s[64][72];
  __shared__ __align__(16) _Float16 wv_s[64][72];
  __shared__ __align__(16) _Float16 xpT_s[64][72];  // [n_local][c]

  const int tid  = threadIdx.x;
  const int w    = tid >> 5, lane = tid & 31;
  const int lg   = lane >> 4, ll = lane & 15;
  const int b    = blockIdx.y;
  const int n0   = blockIdx.x * 64;

  for (int i = tid; i < 64 * 64; i += 128) {
    int o = i >> 6, c = i & 63;
    wq_s[o][c] = (_Float16)qw[o * 64 + c];
    wk_s[o][c] = (_Float16)kw[o * 64 + c];
    wv_s[o][c] = (_Float16)vw[o * 64 + c];
  }

  v8f acc[4];
  #pragma unroll
  for (int t = 0; t < 4; ++t) { v8f z = {}; acc[t] = z; }

  // ---- GEMM1: xproj[c][n] = sum_ci cw[c][ci] * x[ci][n], K=256 in 4 blocks
  for (int cib = 0; cib < 4; ++cib) {
    __syncthreads();
    for (int i = tid; i < 64 * 64; i += 128) {
      int c = i >> 6, cil = i & 63;
      cw_s[c][cil] = (_Float16)cw[c * CIN_ + cib * 64 + cil];
    }
    for (int i = tid; i < 64 * 64; i += 128) {
      int cil = i >> 6, nl = i & 63;
      xT_s[nl][cil] = (_Float16)x[(size_t)(b * CIN_ + cib * 64 + cil) * NN + n0 + nl];
    }
    __syncthreads();
    #pragma unroll
    for (int ks = 0; ks < 2; ++ks) {
      v16h bf = ld_frag_b(&xT_s[w * 16 + ll][0], ks * 32 + 16 * lg);
      #pragma unroll
      for (int ct = 0; ct < 4; ++ct) {
        v16h af = ld_frag_a(&cw_s[ct * 16 + ll][0], ks * 32 + 16 * lg);
        acc[ct] = wmma16(af, bf, acc[ct]);
      }
    }
  }

  // ---- Epilogue 1: bias, write xproj fp32 (coalesced), xpT f16 to LDS
  const int nl = w * 16 + ll;
  #pragma unroll
  for (int ct = 0; ct < 4; ++ct) {
    #pragma unroll
    for (int r = 0; r < 8; ++r) {
      int c = ct * 16 + lg * 8 + r;
      float v = acc[ct][r] + cb[c];
      xproj[(size_t)(b * CC + c) * NN + n0 + nl] = v;
      xpT_s[nl][c] = (_Float16)v;
    }
  }
  __syncthreads();

  // ---- GEMM2: q,k as [n][c_out] tiles; v as [c_out][n] tiles. K=c=64.
  v8f qa[4], ka[4], va[4];
  #pragma unroll
  for (int t = 0; t < 4; ++t) { v8f z = {}; qa[t] = z; ka[t] = z; va[t] = z; }

  #pragma unroll
  for (int ks = 0; ks < 2; ++ks) {
    v16h axp = ld_frag_a(&xpT_s[nl][0], ks * 32 + 16 * lg);   // A for q/k (M=n)
    v16h bxp = ld_frag_b(&xpT_s[nl][0], ks * 32 + 16 * lg);  // B for v (N=n)
    #pragma unroll
    for (int ct = 0; ct < 4; ++ct) {
      v16h bq = ld_frag_b(&wq_s[ct * 16 + ll][0], ks * 32 + 16 * lg);
      qa[ct] = wmma16(axp, bq, qa[ct]);
      v16h bk = ld_frag_b(&wk_s[ct * 16 + ll][0], ks * 32 + 16 * lg);
      ka[ct] = wmma16(axp, bk, ka[ct]);
      v16h av = ld_frag_a(&wv_s[ct * 16 + ll][0], ks * 32 + 16 * lg);
      va[ct] = wmma16(av, bxp, va[ct]);
    }
  }

  #pragma unroll
  for (int ct = 0; ct < 4; ++ct) {
    int o = ct * 16 + ll;
    float qbb = qb[o], kbb = kb[o];
    #pragma unroll
    for (int r = 0; r < 8; ++r) {
      int n = n0 + w * 16 + lg * 8 + r;
      qT[(size_t)(b * NN + n) * CC + o] = (_Float16)(qa[ct][r] + qbb);
      kT[(size_t)(b * NN + n) * CC + o] = (_Float16)(ka[ct][r] + kbb);
    }
    #pragma unroll
    for (int r = 0; r < 8; ++r) {
      int oc = ct * 16 + lg * 8 + r;
      vC[(size_t)(b * CC + oc) * NN + n0 + nl] = (_Float16)(va[ct][r] + vb[oc]);
    }
  }
}

// =====================================================================
// Kernel 2: gating branch. mean over N, two center-tap 3x3 matvecs,
// BN(eval, mean=0,var=1) + relu / sigmoid -> x0[b][c]
// =====================================================================
__global__ __launch_bounds__(256)
void ca_gate_kernel(const float* __restrict__ xproj,
                    const float* __restrict__ c1w, const float* __restrict__ c1b,
                    const float* __restrict__ g1,  const float* __restrict__ b1,
                    const float* __restrict__ c2w, const float* __restrict__ c2b,
                    const float* __restrict__ g2,  const float* __restrict__ b2,
                    float* __restrict__ x0)
{
  __shared__ float mean_s[64];
  __shared__ float h_s[64];
  const int b = blockIdx.x, tid = threadIdx.x;
  const int c = tid >> 2, sub = tid & 3;

  const float* row = xproj + (size_t)(b * CC + c) * NN;
  float s = 0.f;
  for (int i = sub; i < NN; i += 4) s += row[i];
  s += __shfl_xor(s, 1);
  s += __shfl_xor(s, 2);
  if (sub == 0) mean_s[c] = s * (1.f / 4096.f);
  __syncthreads();

  const float bnscale = rsqrtf(1.f + 1e-5f);
  if (tid < 64) {
    float y = c1b[tid];
    for (int i = 0; i < 64; ++i) y += c1w[(tid * 64 + i) * 9 + 4] * mean_s[i];
    y = y * (g1[tid] * bnscale) + b1[tid];
    h_s[tid] = fmaxf(y, 0.f);
  }
  __syncthreads();
  if (tid < 64) {
    float y = c2b[tid];
    for (int i = 0; i < 64; ++i) y += c2w[(tid * 64 + i) * 9 + 4] * h_s[i];
    y = y * (g2[tid] * bnscale) + b2[tid];
    x0[b * CC + tid] = 1.f / (1.f + __expf(-y));
  }
}

// =====================================================================
// Kernel 3: fused flash-attention + epilogue gamma*out + xp*(1+x0)
// 4 waves/block; wave owns 16 query rows; K/V chunks of 64 staged in LDS
// by the Tensor Data Mover, double-buffered (TDM chunk j+1 overlaps WMMA j).
// =====================================================================
__global__ __launch_bounds__(128)
void ca_attn_kernel(const _Float16* __restrict__ qT, const _Float16* __restrict__ kT,
                    const _Float16* __restrict__ vC, const float* __restrict__ xproj,
                    const float* __restrict__ x0,    const float* __restrict__ gamma,
                    float* __restrict__ out)
{
  __shared__ __align__(16) _Float16 kt_s[2][64][72];   // [buf][m_local][c]
  __shared__ __align__(16) _Float16 vc_s[2][64][72];   // [buf][c][m_local]
  __shared__ __align__(16) _Float16 p_s[4][16][72];    // per-wave P tile [n][m_local]
  __shared__ __align__(16) float    ot_s[4][16][68];   // per-wave out tile [n][c]

  const int tid = threadIdx.x;
  const int w   = tid >> 5, lane = tid & 31;
  const int lg  = lane >> 4, ll = lane & 15;
  const int b   = blockIdx.y;
  const int n0  = blockIdx.x * 64;
  const int qrow = n0 + w * 16 + ll;

  // warm the L2 for the fp32 residual tile consumed in the epilogue
  __builtin_prefetch(xproj + (size_t)(b * CC + (tid >> 1)) * NN + n0 + (tid & 1) * 32, 0, 1);

  // Q A-fragments (held in registers for the whole kernel)
  v16h qa[2];
  {
    const _Float16* qp = qT + (size_t)(b * NN + qrow) * CC;
    #pragma unroll
    for (int f = 0; f < 2; ++f) qa[f] = ld_frag_a(qp, f * 32 + 16 * lg);
  }

  v8f oacc[4];
  #pragma unroll
  for (int t = 0; t < 4; ++t) { v8f z = {}; oacc[t] = z; }
  float m_i[8], l_i[8];
  #pragma unroll
  for (int r = 0; r < 8; ++r) { m_i[r] = -1e30f; l_i[r] = 0.f; }

  // TDM: issue chunk 0 into buffer 0 (one issuing wave; EXEC is ignored by TDM)
  if (w == 0) {
    tdm_load_2d_f16(kT + (size_t)b * NN * CC,
                    (unsigned)(uintptr_t)&kt_s[0][0][0], 64u, 64u, 64u, 64u, 64u);
    tdm_load_2d_f16(vC + (size_t)b * CC * NN,
                    (unsigned)(uintptr_t)&vc_s[0][0][0], 64u, 64u, (unsigned)NN, 64u, 64u);
  }

  for (int jc = 0; jc < 64; ++jc) {
    const int buf = jc & 1;
    if (w == 0) __builtin_amdgcn_s_wait_tensorcnt(0);  // chunk jc landed in LDS
    __syncthreads();                                    // publish to all waves
    if (w == 0 && jc + 1 < 64) {                        // overlap DMA of jc+1
      const int m1 = (jc + 1) * 64;
      tdm_load_2d_f16(kT + ((size_t)b * NN + m1) * CC,
                      (unsigned)(uintptr_t)&kt_s[buf ^ 1][0][0], 64u, 64u, 64u, 64u, 64u);
      tdm_load_2d_f16(vC + (size_t)b * CC * NN + m1,
                      (unsigned)(uintptr_t)&vc_s[buf ^ 1][0][0], 64u, 64u, (unsigned)NN, 64u, 64u);
    }

    // S = Q^T K  (D tile [n][m], K over c=64)
    v8f sacc[4];
    #pragma unroll
    for (int t = 0; t < 4; ++t) { v8f z = {}; sacc[t] = z; }
    #pragma unroll
    for (int ks = 0; ks < 2; ++ks) {
      #pragma unroll
      for (int mt = 0; mt < 4; ++mt) {
        v16h bf = ld_frag_b(&kt_s[buf][mt * 16 + ll][0], ks * 32 + 16 * lg);
        sacc[mt] = wmma16(qa[ks], bf, sacc[mt]);
      }
    }

    // online softmax: lane group (16 lanes) covers all 16 columns of a tile;
    // lane<16 holds rows 0..7, lane>=16 rows 8..15.
    float scale[8];
    #pragma unroll
    for (int r = 0; r < 8; ++r) {
      float v = fmaxf(fmaxf(sacc[0][r], sacc[1][r]), fmaxf(sacc[2][r], sacc[3][r]));
      v = fmaxf(v, __shfl_xor(v, 1));
      v = fmaxf(v, __shfl_xor(v, 2));
      v = fmaxf(v, __shfl_xor(v, 4));
      v = fmaxf(v, __shfl_xor(v, 8));
      float mnew = fmaxf(m_i[r], v);
      scale[r] = __expf(m_i[r] - mnew);
      float rsum = 0.f;
      #pragma unroll
      for (int mt = 0; mt < 4; ++mt) {
        float p = __expf(sacc[mt][r] - mnew);
        rsum += p;
        p_s[w][lg * 8 + r][mt * 16 + ll] = (_Float16)p;
      }
      rsum += __shfl_xor(rsum, 1);
      rsum += __shfl_xor(rsum, 2);
      rsum += __shfl_xor(rsum, 4);
      rsum += __shfl_xor(rsum, 8);
      l_i[r] = l_i[r] * scale[r] + rsum;
      m_i[r] = mnew;
    }
    #pragma unroll
    for (int ct = 0; ct < 4; ++ct)
      #pragma unroll
      for (int r = 0; r < 8; ++r)
        oacc[ct][r] *= scale[r];

    // same-wave LDS RAW on p_s (cross-lane) -> drain DS counter
    asm volatile("s_wait_dscnt 0" ::: "memory");

    // O += P V^T  (D tile [n][c], K over m=64)
    #pragma unroll
    for (int ks = 0; ks < 2; ++ks) {
      v16h pf = ld_frag_a(&p_s[w][ll][0], ks * 32 + 16 * lg);
      #pragma unroll
      for (int ct = 0; ct < 4; ++ct) {
        v16h bf = ld_frag_b(&vc_s[buf][ct * 16 + ll][0], ks * 32 + 16 * lg);
        oacc[ct] = wmma16(pf, bf, oacc[ct]);
      }
    }
  }

  // finalize: 1/l, stage [n][c] tile to LDS, then coalesced epilogue store
  #pragma unroll
  for (int ct = 0; ct < 4; ++ct) {
    #pragma unroll
    for (int r = 0; r < 8; ++r)
      ot_s[w][lg * 8 + r][ct * 16 + ll] = oacc[ct][r] / l_i[r];
  }
  asm volatile("s_wait_dscnt 0" ::: "memory");

  const float gm = gamma[0];
  #pragma unroll 4
  for (int i = 0; i < 32; ++i) {
    int c = lg + i * 2;
    float v = ot_s[w][ll][c];
    size_t gi = (size_t)(b * CC + c) * NN + n0 + w * 16 + ll;
    out[gi] = gm * v + xproj[gi] * (1.f + x0[b * CC + c]);
  }
}

// =====================================================================
extern "C" void kernel_launch(void* const* d_in, const int* in_sizes, int n_in,
                              void* d_out, int out_size, void* d_ws, size_t ws_size,
                              hipStream_t stream)
{
  (void)in_sizes; (void)n_in; (void)out_size; (void)ws_size;

  const float* x    = (const float*)d_in[0];
  const float* cw   = (const float*)d_in[1];
  const float* cb   = (const float*)d_in[2];
  const float* qw   = (const float*)d_in[3];
  const float* qb   = (const float*)d_in[4];
  const float* kw   = (const float*)d_in[5];
  const float* kb   = (const float*)d_in[6];
  const float* vw   = (const float*)d_in[7];
  const float* vb   = (const float*)d_in[8];
  const float* gmma = (const float*)d_in[9];
  const float* c1w  = (const float*)d_in[10];
  const float* c1b  = (const float*)d_in[11];
  const float* g1   = (const float*)d_in[12];
  const float* b1   = (const float*)d_in[13];
  const float* c2w  = (const float*)d_in[14];
  const float* c2b  = (const float*)d_in[15];
  const float* g2   = (const float*)d_in[16];
  const float* b2   = (const float*)d_in[17];

  const size_t elems = (size_t)BB * CC * NN;     // 2,097,152
  float*    xproj = (float*)d_ws;                //  8 MB fp32
  _Float16* qT    = (_Float16*)(xproj + elems);  //  4 MB f16 [B][N][C]
  _Float16* kT    = qT + elems;                  //  4 MB f16 [B][N][C]
  _Float16* vC    = kT + elems;                  //  4 MB f16 [B][C][N]
  float*    x0    = (float*)(vC + elems);        //  2 KB

  ca_proj_kernel<<<dim3(NN / 64, BB), 128, 0, stream>>>(
      x, cw, cb, qw, qb, kw, kb, vw, vb, xproj, qT, kT, vC);
  ca_gate_kernel<<<dim3(BB), 256, 0, stream>>>(
      xproj, c1w, c1b, g1, b1, c2w, c2b, g2, b2, x0);
  ca_attn_kernel<<<dim3(NN / 64, BB), 128, 0, stream>>>(
      qT, kT, vC, xproj, x0, gmma, (float*)d_out);
}
